// BiTreeLSTM_Backward_83099027243634
// MI455X (gfx1250) — matrix operand order repaired and checked by reading, not compiled, not run on hardware
//
#include <hip/hip_runtime.h>
#include <hip/hip_bf16.h>

// ---------------------------------------------------------------------------
// TreeLSTM forward for MI455X (gfx1250, wave32, WMMA + TDM).
//   Phase 1: xproj = features @ Wx^T + (bx+bh)      [8192x2048]x[2048x4096]
//   Phase 2: per tree level: iofu = xproj + h_par @ Wh^T ; LSTM cell.
// bf16 WMMA operands, fp32 accumulate, fp32 states/activations.
// B panels staged into LDS by the Tensor Data Mover (double-buffered,
// s_wait_tensorcnt); A fragments register double-buffered global loads
// (phase-2 A rows are parent-gathers, not TDM-tileable).
// ---------------------------------------------------------------------------

#define N_NODES 8192
#define FEAT    2048
#define HDIM    1024
#define FOURH   4096

typedef __bf16 bf16_t;
typedef bf16_t   bf16x8  __attribute__((ext_vector_type(8)));
typedef bf16_t   bf16x16 __attribute__((ext_vector_type(16)));
typedef float    floatx8 __attribute__((ext_vector_type(8)));
typedef unsigned u32x4   __attribute__((ext_vector_type(4)));
typedef int      i32x4   __attribute__((ext_vector_type(4)));
typedef int      i32x8   __attribute__((ext_vector_type(8)));

// Build a 16-element bf16 fragment from two contiguous 8-element (16B) chunks.
__device__ __forceinline__ bf16x16 ld_split(const bf16_t* p0, const bf16_t* p1) {
    bf16x8 lo = *(const bf16x8*)p0;
    bf16x8 hi = *(const bf16x8*)p1;
    return __builtin_shufflevector(lo, hi, 0, 1, 2, 3, 4, 5, 6, 7,
                                           8, 9, 10, 11, 12, 13, 14, 15);
}

// ---------------------------------------------------------------------------
// TDM: load a [128 rows x 32 K-elements] bf16 tile of B (row-major, row
// stride K elements) into LDS at byte offset lds_off. 2D descriptor:
//   group0: count=1 | lds_addr | global_addr | type=2
//   group1: data_size=2B, tensor_dim0=K, tensor_dim1=4096, tile=32x128,
//           tensor_dim0_stride=K
// ---------------------------------------------------------------------------
__device__ __forceinline__ void tdm_load_b_tile(const bf16_t* gptr,
                                                unsigned lds_off, int K) {
    unsigned long long ga = (unsigned long long)(uintptr_t)gptr;
    u32x4 g0;
    g0[0] = 1u;                                        // count=1, user desc
    g0[1] = lds_off;                                   // lds_addr (bytes)
    g0[2] = (unsigned)(ga & 0xFFFFFFFFu);              // global_addr[31:0]
    g0[3] = (unsigned)((ga >> 32) & 0x01FFFFFFu) | (2u << 30); // addr[56:32]|type=2
    i32x8 g1;
    g1[0] = 0x00010000;                                // data_size=1 (2 bytes)
    g1[1] = (int)(((unsigned)K & 0xFFFFu) << 16);      // tensor_dim0[15:0]
    g1[2] = (int)((4096u << 16) | (((unsigned)K >> 16) & 0xFFFFu));
                                                       // tensor_dim1[15:0]|dim0[31:16]
    g1[3] = (int)(32u << 16);                          // tile_dim0=32 (K elems)
    g1[4] = 128;                                       // tile_dim1=128 rows
    g1[5] = K;                                         // tensor_dim0_stride[31:0]
    g1[6] = 0;
    g1[7] = 0;
    i32x4 z4 = {0, 0, 0, 0};
#if __clang_major__ >= 23
    i32x8 z8 = {0, 0, 0, 0, 0, 0, 0, 0};
    __builtin_amdgcn_tensor_load_to_lds(g0, g1, z4, z4, z8, 0);
#else
    __builtin_amdgcn_tensor_load_to_lds(g0, g1, z4, z4, 0);
#endif
}

// ---------------------------------------------------------------------------
// GEMM: C[M x 4096] (+)= A[M x K] * B^T, B row-major [4096 x K], so a WMMA
// B-fragment column == contiguous slice of a B row (no transpose anywhere).
// mode 0: C = acc + bias[col]   (phase 1, A row r = global row r)
// mode 1: C += acc              (phase 2, A row r gathered via parent index)
// Block: 256 thr = 8 waves as 4(M) x 2(N); block tile 64 x 128; K step 32.
// Each wave: 1 A frag (global, register double-buffered), 4 B frags (LDS,
// loaded as one clause), 4 WMMAs per K step.  B panel TDM double-buffered.
// ---------------------------------------------------------------------------
__global__ __launch_bounds__(256)
void gemm_bf16_nt(const bf16_t* __restrict__ A, const bf16_t* __restrict__ B,
                  const float* __restrict__ bias, float* __restrict__ C,
                  const int* __restrict__ parent, int t0, int Mrows, int K,
                  int mode) {
    __shared__ __align__(16) bf16_t ldsB[2][128 * 32];

    const int lane = threadIdx.x & 31;
    const int wave = threadIdx.x >> 5;
    const int wm   = wave & 3;        // 0..3  (M sub-tile)
    const int wn   = wave >> 2;       // 0..1  (N sub-tile, 64 cols each)
    const int l15  = lane & 15;
    const int hi   = lane >> 4;       // lane half

    // --- A row pointer (optionally gathered through parent index) ---
    int r  = blockIdx.x * 64 + wm * 16 + l15;
    int rr = (r < Mrows) ? r : (Mrows - 1);
    const bf16_t* arow;
    if (mode == 1) {
        int p = parent[t0 + rr];           // root parent = -1 -> slot 0
        arow = A + (size_t)(p + 1) * (size_t)K;
    } else {
        arow = A + (size_t)rr * (size_t)K;
    }
    const int kA = hi * 16;  // A frag: lanes0-15 K{0..15}; lanes16-31 {16..31}

    const int colBase = blockIdx.y * 128;
    const unsigned lds_base = (unsigned)(uintptr_t)(void*)&ldsB[0][0];
    const bf16_t* Bpanel = B + (size_t)colBase * (size_t)K;

    floatx8 acc[4];
#pragma unroll
    for (int nt = 0; nt < 4; ++nt)
#pragma unroll
        for (int v = 0; v < 8; ++v) acc[nt][v] = 0.f;

    const int nsteps = K >> 5;

    // Prime pipelines: first TDM B tile, first A fragment.
    if (wave == 0) tdm_load_b_tile(Bpanel, lds_base, K);
    bf16x16 a_cur = ld_split(arow + kA, arow + kA + 8);

    // Per-lane base offset into a 128x32 LDS tile (row-major, 32 elems/row).
    const int boff = (wn * 64 + l15) * 32 + hi * 16;

    for (int ks = 0; ks < nsteps; ++ks) {
        const int k = ks << 5;
        if (wave == 0) {
            if (ks + 1 < nsteps) {
                tdm_load_b_tile(Bpanel + (k + 32),
                                lds_base + (unsigned)(((ks + 1) & 1) * (128 * 32 * 2)),
                                K);
                __builtin_amdgcn_s_wait_tensorcnt(1);  // current tile landed
            } else {
                __builtin_amdgcn_s_wait_tensorcnt(0);  // last tile landed
            }
        }
        __syncthreads();                               // tile visible to all waves

        // All B fragments up front -> one ds clause, one wait, 4 WMMAs.
        const bf16_t* bt = &ldsB[ks & 1][0];
        bf16x16 b0 = ld_split(bt + boff,             bt + boff + 8);
        bf16x16 b1 = ld_split(bt + boff + 16 * 32,   bt + boff + 16 * 32 + 8);
        bf16x16 b2 = ld_split(bt + boff + 32 * 32,   bt + boff + 32 * 32 + 8);
        bf16x16 b3 = ld_split(bt + boff + 48 * 32,   bt + boff + 48 * 32 + 8);

        // Prefetch next A fragment during the WMMA chain (clamped at the end).
        const int kn = (ks + 1 < nsteps) ? (k + 32) : k;
        bf16x16 a_nxt = ld_split(arow + kn + kA, arow + kn + kA + 8);
        __builtin_prefetch(arow + k + 256, 0, 1);      // global_prefetch_b8

        acc[0] = __builtin_amdgcn_wmma_f32_16x16x32_bf16(
            false, a_cur, false, b0, (short)0, acc[0], false, false);
        acc[1] = __builtin_amdgcn_wmma_f32_16x16x32_bf16(
            false, a_cur, false, b1, (short)0, acc[1], false, false);
        acc[2] = __builtin_amdgcn_wmma_f32_16x16x32_bf16(
            false, a_cur, false, b2, (short)0, acc[2], false, false);
        acc[3] = __builtin_amdgcn_wmma_f32_16x16x32_bf16(
            false, a_cur, false, b3, (short)0, acc[3], false, false);

        a_cur = a_nxt;
        __syncthreads();   // everyone done reading before TDM overwrites buffer
    }

    // C/D layout: lane -> N col (lane&15); VGPR v -> M row (lanes16-31: +8)
    const int mBase = blockIdx.x * 64 + wm * 16 + hi * 8;
#pragma unroll
    for (int nt = 0; nt < 4; ++nt) {
        const int col = colBase + wn * 64 + nt * 16 + l15;
#pragma unroll
        for (int v = 0; v < 8; ++v) {
            int m = mBase + v;
            if (m < Mrows) {
                size_t idx = (size_t)m * FOURH + col;
                if (mode == 1) C[idx] += acc[nt][v];
                else           C[idx] = acc[nt][v] + bias[col];
            }
        }
    }
}

// ---------------------------------------------------------------------------
// LSTM cell for one tree level. One block per node, 256 threads x 4 columns.
// iofu layout per row: [i | o | f | u] each HDIM wide.
// ---------------------------------------------------------------------------
__global__ __launch_bounds__(256)
void lstm_cell(const float* __restrict__ iofu, const int* __restrict__ parent,
               int t0, float* __restrict__ c_buf, bf16_t* __restrict__ h_bf,
               float* __restrict__ out) {
    const int t = t0 + blockIdx.x;
    const int p = parent[t];
    const float* row   = iofu + (size_t)blockIdx.x * FOURH;
    const float* cprow = c_buf + (size_t)(p + 1) * HDIM;
    const size_t wbase = (size_t)(t + 1) * HDIM;
#pragma unroll
    for (int j = 0; j < HDIM / 256; ++j) {
        int col = threadIdx.x + j * 256;
        float iv = row[col];
        float ov = row[HDIM + col];
        float fv = row[2 * HDIM + col];
        float uv = row[3 * HDIM + col];
        iv = 1.f / (1.f + __expf(-iv));
        ov = 1.f / (1.f + __expf(-ov));
        fv = 1.f / (1.f + __expf(-fv));
        uv = tanhf(uv);
        float cv = iv * uv + fv * cprow[col];
        float hv = ov * tanhf(cv);
        c_buf[wbase + col] = cv;
        h_bf[wbase + col]  = (bf16_t)hv;
        out[(size_t)t * HDIM + col] = hv;
    }
}

// ---------------------------------------------------------------------------
// Prep kernels.
// ---------------------------------------------------------------------------
__global__ void cast_f32_to_bf16(const float* __restrict__ s,
                                 bf16_t* __restrict__ d, size_t n) {
    size_t i = (size_t)blockIdx.x * blockDim.x + threadIdx.x;
    size_t stride = (size_t)gridDim.x * blockDim.x;
    for (; i < n; i += stride) d[i] = (bf16_t)s[i];
}

__global__ void add_bias(const float* __restrict__ bx, const float* __restrict__ bh,
                         float* __restrict__ bxh) {
    int i = blockIdx.x * blockDim.x + threadIdx.x;
    bxh[i] = bx[i] + bh[i];
}

__global__ void init_roots(const float* __restrict__ root_h,
                           const float* __restrict__ root_c,
                           float* __restrict__ c_buf, bf16_t* __restrict__ h_bf) {
    int i = blockIdx.x * blockDim.x + threadIdx.x;
    c_buf[i] = root_c[i];
    h_bf[i]  = (bf16_t)root_h[i];
}

// ---------------------------------------------------------------------------
// Host launcher.
// Inputs: 0=features 1=Wx 2=bx 3=Wh 4=bh 5=parent_idx 6=root_c 7=root_h
// ---------------------------------------------------------------------------
extern "C" void kernel_launch(void* const* d_in, const int* in_sizes, int n_in,
                              void* d_out, int out_size, void* d_ws, size_t ws_size,
                              hipStream_t stream) {
    const float* features = (const float*)d_in[0];
    const float* Wx       = (const float*)d_in[1];
    const float* bx       = (const float*)d_in[2];
    const float* Wh       = (const float*)d_in[3];
    const float* bh       = (const float*)d_in[4];
    const int*   parent   = (const int*)d_in[5];
    const float* root_c   = (const float*)d_in[6];
    const float* root_h   = (const float*)d_in[7];
    float* out = (float*)d_out;
    (void)in_sizes; (void)n_in; (void)out_size; (void)ws_size;

    char* ws = (char*)d_ws;
    size_t off = 0;
    auto take = [&](size_t bytes) -> char* {
        char* p = ws + off;
        off = (off + bytes + 255) & ~(size_t)255;
        return p;
    };
    bf16_t* feat_bf = (bf16_t*)take((size_t)N_NODES * FEAT * 2);   // 32 MB
    bf16_t* wx_bf   = (bf16_t*)take((size_t)FOURH * FEAT * 2);     // 16 MB
    bf16_t* wh_bf   = (bf16_t*)take((size_t)FOURH * HDIM * 2);     //  8 MB
    float*  bxh     = (float*) take((size_t)FOURH * 4);
    float*  xproj   = (float*) take((size_t)N_NODES * FOURH * 4);  // 128 MB
    float*  c_buf   = (float*) take((size_t)(N_NODES + 1) * HDIM * 4); // 32 MB
    bf16_t* h_bf    = (bf16_t*)take((size_t)(N_NODES + 1) * HDIM * 2); // 16 MB

    // --- prep: casts, bias sum, root state ---
    cast_f32_to_bf16<<<2048, 256, 0, stream>>>(features, feat_bf,
                                               (size_t)N_NODES * FEAT);
    cast_f32_to_bf16<<<1024, 256, 0, stream>>>(Wx, wx_bf, (size_t)FOURH * FEAT);
    cast_f32_to_bf16<<<512, 256, 0, stream>>>(Wh, wh_bf, (size_t)FOURH * HDIM);
    add_bias<<<FOURH / 256, 256, 0, stream>>>(bx, bh, bxh);
    init_roots<<<HDIM / 256, 256, 0, stream>>>(root_h, root_c, c_buf, h_bf);

    // --- phase 1: xproj = features @ Wx^T + (bx+bh) ---
    gemm_bf16_nt<<<dim3(N_NODES / 64, FOURH / 128), 256, 0, stream>>>(
        feat_bf, wx_bf, bxh, xproj, nullptr, 0, N_NODES, FEAT, 0);

    // --- phase 2: level-parallel recurrence (stream order = level order) ---
    for (int lvl = 0;; ++lvl) {
        int t0 = (1 << lvl) - 1;
        if (t0 >= N_NODES) break;
        int t1 = (2 << lvl) - 1;
        if (t1 > N_NODES) t1 = N_NODES;
        int L = t1 - t0;

        dim3 g((L + 63) / 64, FOURH / 128);
        gemm_bf16_nt<<<g, 256, 0, stream>>>(
            h_bf, wh_bf, nullptr, xproj + (size_t)t0 * FOURH,
            parent, t0, L, HDIM, 1);

        lstm_cell<<<L, 256, 0, stream>>>(
            xproj + (size_t)t0 * FOURH, parent, t0, c_buf, h_bf, out);
    }
}
